// BitLinear_47794396070042
// MI455X (gfx1250) — matrix operand order repaired and checked by reading, not compiled, not run on hardware
//
#include <hip/hip_runtime.h>

// ---- CDNA5 (gfx1250) BitLinear: ternary-weight GEMM via v_wmma_f32_16x16x32_bf16
// out[b,s,o] = scale[o] * sum_k x[b,s,k] * clip(round(w[o,k]/0.5),-1,1)
// GEMM: M=8192 (B*S), N=11008 (OUT_F), K=4096 (IN_F)
//
// Fast path (needs ws >= 157.3 MB):
//   1) quant_x_k : x fp32 -> bf16                [M][K]      (67.1 MB)
//   2) quant_w_t : w fp32 -> ternary bf16, T     [K][N]      (90.2 MB)
//   3) gemm_bf16_wmma : conversion-free double-buffered WMMA GEMM,
//      staging via GLOBAL_LOAD_ASYNC_TO_LDS_B128 (ASYNCcnt) when available.
// Fallback: fused single-kernel version.

typedef __attribute__((ext_vector_type(16))) __bf16        v16bf;
typedef __attribute__((ext_vector_type(8)))  float         v8f;
typedef __attribute__((ext_vector_type(4)))  float         v4f;
typedef __attribute__((ext_vector_type(4)))  unsigned int  v4u;
typedef __attribute__((ext_vector_type(2)))  unsigned int  v2u;
typedef __attribute__((ext_vector_type(4)))  int           v4i;

#define IN_F   4096
#define OUT_F  11008
#define M_TOT  8192

#define BM 128
#define BN 128
#define BK 32
#define ASTR 40    // BK + 8 pad (ushorts): 80B rows -> conflict-free A fragment reads
#define BSTR 136   // BN + 8 pad (ushorts): 272B rows, 16B-aligned B fragment reads

#if __has_builtin(__builtin_amdgcn_global_load_async_to_lds_b128)
#define HAVE_ASYNC_LDS 1
// signature (from clang diagnostic): (v4i AS1*, v4i AS3*, imm offset, imm cpol);
// the immediate offset applies to BOTH global and LDS addresses (ISA §10.7/15.18)
#define ASYNC_CP16(g, l, off)                                              \
    __builtin_amdgcn_global_load_async_to_lds_b128(                        \
        (__attribute__((address_space(1))) v4i*)(g),                       \
        (__attribute__((address_space(3))) v4i*)(l), (off), 0)
#define WAIT_ASYNC() asm volatile("s_wait_asynccnt 0x0" ::: "memory")
#else
#define HAVE_ASYNC_LDS 0
#endif

struct Pack32 { v4u lo, hi; };   // 32 bytes == v16bf

__device__ __forceinline__ unsigned int f2bf(float f) {
    // round-to-nearest-even fp32 -> bf16 (low 16 bits)
    unsigned int u = __builtin_bit_cast(unsigned int, f);
    u += 0x7FFFu + ((u >> 16) & 1u);
    return u >> 16;
}

__device__ __forceinline__ float ternary(float w) {
    float t = __builtin_rintf(w * 2.0f);              // round(w / 0.5), RNE
    return fminf(1.0f, fmaxf(-1.0f, t));              // clip to {-1,0,1}
}

// ============================ phase 1a: x -> bf16 ============================
__global__ __launch_bounds__(256)
void quant_x_k(const float* __restrict__ x, unsigned short* __restrict__ xq) {
    const size_t i = ((size_t)blockIdx.x * 256 + threadIdx.x) * 8;
    v4f a = *(const v4f*)(x + i);
    v4f b = *(const v4f*)(x + i + 4);
    v4u d;
    d.x = f2bf(a.x) | (f2bf(a.y) << 16);
    d.y = f2bf(a.z) | (f2bf(a.w) << 16);
    d.z = f2bf(b.x) | (f2bf(b.y) << 16);
    d.w = f2bf(b.z) | (f2bf(b.w) << 16);
    *(v4u*)(xq + i) = d;
}

// ================= phase 1b: w -> ternary bf16, transposed ==================
__global__ __launch_bounds__(256)
void quant_w_t(const float* __restrict__ w, unsigned short* __restrict__ wqT) {
    __shared__ unsigned short tile[32][36];
    const int n0 = blockIdx.x * 32;
    const int k0 = blockIdx.y * 32;
    const int t  = threadIdx.x;
    {   // coalesced read along k
        const int r = t >> 3;           // n within tile
        const int c = (t & 7) << 2;     // k within tile
        v4f v = *(const v4f*)(w + (size_t)(n0 + r) * IN_F + k0 + c);
#pragma unroll
        for (int j = 0; j < 4; ++j)
            tile[r][c + j] = (unsigned short)f2bf(ternary(v[j]));
    }
    __syncthreads();
    {   // coalesced write along n
        const int kk = t >> 3;
        const int nn = (t & 7) << 2;
        v2u d;
        d.x = (unsigned int)tile[nn + 0][kk] | ((unsigned int)tile[nn + 1][kk] << 16);
        d.y = (unsigned int)tile[nn + 2][kk] | ((unsigned int)tile[nn + 3][kk] << 16);
        *(v2u*)(wqT + (size_t)(k0 + kk) * OUT_F + n0 + nn) = d;
    }
}

// ==================== phase 2: conversion-free WMMA GEMM ====================
__global__ __launch_bounds__(256)
void gemm_bf16_wmma(const unsigned short* __restrict__ xq,   // [M][K] bf16
                    const unsigned short* __restrict__ wqT,  // [K][N] bf16 ternary
                    const float* __restrict__ scale,
                    float* __restrict__ out)
{
    __shared__ alignas(16) unsigned short As[2][BM][ASTR];   // [m][k]
    __shared__ alignas(16) unsigned short Bs[2][BK][BSTR];   // [k][n]

    const int tid  = threadIdx.x;
    const int lane = tid & 31;
    const int wave = tid >> 5;
    const int wm   = wave & 1;      // 2 waves along M
    const int wn   = wave >> 1;     // 4 waves along N

    const int n_blk = blockIdx.x * BN;
    const int m_blk = blockIdx.y * BM;

    // loop-invariant 32-bit element offsets (k0 term is uniform -> SALU)
    const unsigned a_row = tid >> 1;                 // 0..127
    const unsigned a_off = (tid & 1) << 4;           // 0 or 16 (bf16 elems)
    const unsigned b_row = tid >> 3;                 // 0..31  (k)
    const unsigned b_off = (tid & 7) << 4;           // 0..112 (n)
    const unsigned a_idx = (unsigned)(m_blk + a_row) * IN_F + a_off;
    const unsigned b_idx = b_row * (unsigned)OUT_F + (unsigned)n_blk + b_off;

    v8f acc[4][2] = {};
    const int a_m  = lane & 15;              // A fragment: lane -> M row
    const int a_kb = (lane >> 4) << 3;       // lanes 16..31 hold K=8..15 / 24..31

    auto compute = [&](int st) {
        v16bf afrag[4], bfrag[2];
#pragma unroll
        for (int im = 0; im < 4; ++im) {
            const unsigned short* p = &As[st][wm * 64 + im * 16 + a_m][0];
            Pack32 t;
            t.lo = *(const v4u*)(p + a_kb);        // elems 0..7  : K = base+0..7
            t.hi = *(const v4u*)(p + 16 + a_kb);   // elems 8..15 : K = base+16..23
            afrag[im] = __builtin_bit_cast(v16bf, t);
        }
#pragma unroll
        for (int in = 0; in < 2; ++in) {
            const unsigned short* p = &Bs[st][a_kb][wn * 32 + in * 16 + a_m];
            unsigned short t[16];
#pragma unroll
            for (int s = 0; s < 8; ++s) {
                t[s]     = p[s * BSTR];            // elems 0..7  : K = base+0..7
                t[8 + s] = p[(16 + s) * BSTR];     // elems 8..15 : K = base+16..23
            }
            bfrag[in] = __builtin_bit_cast(v16bf, t);
        }
#pragma unroll
        for (int im = 0; im < 4; ++im)
#pragma unroll
            for (int in = 0; in < 2; ++in)
                acc[im][in] = __builtin_amdgcn_wmma_f32_16x16x32_bf16(
                    false, afrag[im], false, bfrag[in],
                    (short)0, acc[im][in], false, false);
    };

#if HAVE_ASYNC_LDS
    // -------- async staging: global -> LDS direct, tracked by ASYNCcnt ------
    auto aload = [&](int st, unsigned k0) {
        const unsigned short* xp = xq + a_idx + k0;
        const unsigned short* wp = wqT + b_idx + k0 * (unsigned)OUT_F;
        unsigned short* la = &As[st][a_row][a_off];
        unsigned short* lb = &Bs[st][b_row][b_off];
        ASYNC_CP16(xp, la, 0);
        ASYNC_CP16(xp, la, 16);
        ASYNC_CP16(wp, lb, 0);
        ASYNC_CP16(wp, lb, 16);
    };

    aload(0, 0);
    WAIT_ASYNC();
    __syncthreads();

    int st = 0;
    for (unsigned k0 = BK; k0 < IN_F; k0 += BK) {
        aload(st ^ 1, k0);                          // next tile -> other buffer
        __builtin_prefetch(xq + a_idx + k0 + BK, 0, 2);            // speculative
        __builtin_prefetch(wqT + b_idx + (k0 + BK) * (unsigned)OUT_F, 0, 2);
        compute(st);                                // WMMA on current buffer
        WAIT_ASYNC();                               // our async writes landed
        __syncthreads();                            // everyone's landed
        st ^= 1;
    }
    compute(st);
#else
    // -------- sync staging fallback: global -> VGPR -> LDS ------------------
    v4u ra0, ra1, rb0, rb1;
    auto gload = [&](unsigned k0) {
        const unsigned short* xp = xq + a_idx + k0;
        const unsigned short* wp = wqT + b_idx + k0 * (unsigned)OUT_F;
        ra0 = *(const v4u*)(xp);
        ra1 = *(const v4u*)(xp + 8);
        rb0 = *(const v4u*)(wp);
        rb1 = *(const v4u*)(wp + 8);
    };
    auto sstore = [&](int st) {
        v4u* pa = (v4u*)&As[st][a_row][a_off];
        pa[0] = ra0; pa[1] = ra1;
        v4u* pb = (v4u*)&Bs[st][b_row][b_off];
        pb[0] = rb0; pb[1] = rb1;
    };

    gload(0);
    sstore(0);
    __syncthreads();

    int st = 0;
    for (unsigned k0 = BK; k0 < IN_F; k0 += BK) {
        gload(k0);
        __builtin_prefetch(xq + a_idx + k0 + BK, 0, 2);
        __builtin_prefetch(wqT + b_idx + (k0 + BK) * (unsigned)OUT_F, 0, 2);
        compute(st);
        sstore(st ^ 1);
        __syncthreads();
        st ^= 1;
    }
    compute(st);
#endif

    // epilogue: apply per-channel scale (C/D: VGPR r -> M=r+8*(lane>=16), N=lane&15)
#pragma unroll
    for (int in = 0; in < 2; ++in) {
        const int n = n_blk + wn * 32 + in * 16 + (lane & 15);
        const float s = scale[n];
#pragma unroll
        for (int im = 0; im < 4; ++im) {
            const int m0 = m_blk + wm * 64 + im * 16 + ((lane >> 4) << 3);
            v8f c = acc[im][in];
#pragma unroll
            for (int r = 0; r < 8; ++r)
                out[(size_t)(m0 + r) * OUT_F + n] = c[r] * s;
        }
    }
}

// ===================== fallback: fused single-pass kernel ====================
__global__ __launch_bounds__(256)
void bitlinear_ternary_wmma(const float* __restrict__ x,
                            const float* __restrict__ w,
                            const float* __restrict__ scale,
                            float* __restrict__ out)
{
    __shared__ alignas(16) unsigned short As[2][BM][ASTR];
    __shared__ alignas(16) unsigned short Bs[2][BK][BSTR];

    const int tid  = threadIdx.x;
    const int lane = tid & 31;
    const int wave = tid >> 5;
    const int wm   = wave & 1;
    const int wn   = wave >> 1;
    const int n_blk = blockIdx.x * BN;
    const int m_blk = blockIdx.y * BM;
    const int lrow  = tid >> 3;
    const int lcol4 = (tid & 7) << 2;

    v4f regA[4], regB[4];
    auto gload = [&](int k0) {
#pragma unroll
        for (int i = 0; i < 4; ++i) {
            const int row = lrow + i * 32;
            regA[i] = *(const v4f*)(x + (size_t)(m_blk + row) * IN_F + k0 + lcol4);
            regB[i] = *(const v4f*)(w + (size_t)(n_blk + row) * IN_F + k0 + lcol4);
        }
    };
    auto sstore = [&](int st) {
#pragma unroll
        for (int i = 0; i < 4; ++i) {
            const int row = lrow + i * 32;
            unsigned int p0 = f2bf(regA[i].x) | (f2bf(regA[i].y) << 16);
            unsigned int p1 = f2bf(regA[i].z) | (f2bf(regA[i].w) << 16);
            unsigned int* da = (unsigned int*)&As[st][row][lcol4];
            da[0] = p0; da[1] = p1;
#pragma unroll
            for (int j = 0; j < 4; ++j)
                Bs[st][lcol4 + j][row] = (unsigned short)f2bf(ternary(regB[i][j]));
        }
    };

    v8f acc[4][2] = {};
    const int a_m  = lane & 15;
    const int a_kb = (lane >> 4) << 3;
    auto compute = [&](int st) {
        v16bf afrag[4], bfrag[2];
#pragma unroll
        for (int im = 0; im < 4; ++im) {
            const unsigned short* p = &As[st][wm * 64 + im * 16 + a_m][0];
            Pack32 t;
            t.lo = *(const v4u*)(p + a_kb);
            t.hi = *(const v4u*)(p + 16 + a_kb);
            afrag[im] = __builtin_bit_cast(v16bf, t);
        }
#pragma unroll
        for (int in = 0; in < 2; ++in) {
            const unsigned short* p = &Bs[st][a_kb][wn * 32 + in * 16 + a_m];
            unsigned short t[16];
#pragma unroll
            for (int s = 0; s < 8; ++s) {
                t[s]     = p[s * BSTR];
                t[8 + s] = p[(16 + s) * BSTR];
            }
            bfrag[in] = __builtin_bit_cast(v16bf, t);
        }
#pragma unroll
        for (int im = 0; im < 4; ++im)
#pragma unroll
            for (int in = 0; in < 2; ++in)
                acc[im][in] = __builtin_amdgcn_wmma_f32_16x16x32_bf16(
                    false, afrag[im], false, bfrag[in],
                    (short)0, acc[im][in], false, false);
    };

    gload(0);
    sstore(0);
    __syncthreads();
    int st = 0;
    for (int k0 = BK; k0 < IN_F; k0 += BK) {
        gload(k0);
        compute(st);
        sstore(st ^ 1);
        __syncthreads();
        st ^= 1;
    }
    compute(st);

#pragma unroll
    for (int in = 0; in < 2; ++in) {
        const int n = n_blk + wn * 32 + in * 16 + (lane & 15);
        const float s = scale[n];
#pragma unroll
        for (int im = 0; im < 4; ++im) {
            const int m0 = m_blk + wm * 64 + im * 16 + ((lane >> 4) << 3);
            v8f c = acc[im][in];
#pragma unroll
            for (int r = 0; r < 8; ++r)
                out[(size_t)(m0 + r) * OUT_F + n] = c[r] * s;
        }
    }
}

// ================================ dispatch ==================================
extern "C" void kernel_launch(void* const* d_in, const int* in_sizes, int n_in,
                              void* d_out, int out_size, void* d_ws, size_t ws_size,
                              hipStream_t stream) {
    const float* x     = (const float*)d_in[0];   // [4, 2048, 4096] fp32
    const float* w     = (const float*)d_in[1];   // [11008, 4096] fp32
    const float* scale = (const float*)d_in[2];   // [11008] fp32
    float* out = (float*)d_out;                   // [4, 2048, 11008] fp32

    const size_t XQ_ELEMS = (size_t)M_TOT * IN_F;     // 33,554,432
    const size_t WQ_ELEMS = (size_t)IN_F * OUT_F;     // 45,088,768
    const size_t need = (XQ_ELEMS + WQ_ELEMS) * sizeof(unsigned short);  // ~157.3 MB

    dim3 grid(OUT_F / BN, M_TOT / BM);            // (86, 64), exact tiling

    if (ws_size >= need) {
        unsigned short* xq  = (unsigned short*)d_ws;
        unsigned short* wqT = xq + XQ_ELEMS;
        quant_x_k<<<(unsigned)(XQ_ELEMS / (8 * 256)), 256, 0, stream>>>(x, xq);
        quant_w_t<<<dim3(OUT_F / 32, IN_F / 32), 256, 0, stream>>>(w, wqT);
        gemm_bf16_wmma<<<grid, dim3(256), 0, stream>>>(xq, wqT, scale, out);
    } else {
        bitlinear_ternary_wmma<<<grid, dim3(256), 0, stream>>>(x, w, scale, out);
    }
}
